// MSARowAttentionWithPairBias_51977694216827
// MI455X (gfx1250) — matrix operand order repaired and checked by reading, not compiled, not run on hardware
//
#include <hip/hip_runtime.h>

typedef __attribute__((ext_vector_type(16))) __bf16 v16bf;
typedef __attribute__((ext_vector_type(8)))  float  v8f;

#define DDIM 256   // model dim
#define LDIM 256   // residues
#define SDIM 128   // sequences
#define NH   8
#define HD   32

// ---------- helpers ----------
static __device__ inline __bf16 f2bf(float f) {
  unsigned u = __builtin_bit_cast(unsigned, f);
  u += 0x7fffu + ((u >> 16) & 1u);          // round-to-nearest-even
  unsigned short s = (unsigned short)(u >> 16);
  return __builtin_bit_cast(__bf16, s);
}
// pack two f32 -> two bf16 in one dword (x -> low half, y -> high half)
static __device__ inline unsigned pack2(float x, float y) {
  unsigned ux = __builtin_bit_cast(unsigned, x);
  unsigned uy = __builtin_bit_cast(unsigned, y);
  ux += 0x7fffu + ((ux >> 16) & 1u);
  uy += 0x7fffu + ((uy >> 16) & 1u);
  return (ux >> 16) | (uy & 0xffff0000u);
}
static __device__ inline float bflo(unsigned u) { return __builtin_bit_cast(float, u << 16); }
static __device__ inline float bfhi(unsigned u) { return __builtin_bit_cast(float, u & 0xffff0000u); }

static __device__ inline v16bf load16g(const __bf16* p) {  // 32B contiguous (global)
  union { v16bf v; uint4 q[2]; } u;
  u.q[0] = ((const uint4*)p)[0];
  u.q[1] = ((const uint4*)p)[1];
  return u.v;
}
static __device__ inline v16bf load8_8(const __bf16* p0, const __bf16* p1) { // 2x16B
  union { v16bf v; uint4 q[2]; } u;
  u.q[0] = *(const uint4*)p0;
  u.q[1] = *(const uint4*)p1;
  return u.v;
}
static __device__ inline v16bf load16l(const __bf16* p) {  // 32B contiguous (LDS)
  union { v16bf v; uint4 q[2]; } u;
  u.q[0] = ((const uint4*)p)[0];
  u.q[1] = ((const uint4*)p)[1];
  return u.v;
}
static __device__ inline v8f wmma_bf16(v16bf a, v16bf b, v8f c) {
  return __builtin_amdgcn_wmma_f32_16x16x32_bf16(false, a, false, b, (short)0, c, false, false);
}

// =====================================================================
// Kernel 0: one-shot msa f32 -> bf16 (packed conversion, 8 elems/thread)
// =====================================================================
__global__ __launch_bounds__(256) void cvt_msa_kernel(
    const float* __restrict__ in, __bf16* __restrict__ out)
{
  size_t i = ((size_t)blockIdx.x * 256 + threadIdx.x) * 8;
  float4 a = *(const float4*)(in + i);
  float4 b = *(const float4*)(in + i + 4);
  uint4 o;
  o.x = pack2(a.x, a.y);
  o.y = pack2(a.z, a.w);
  o.z = pack2(b.x, b.y);
  o.w = pack2(b.z, b.w);
  *(uint4*)(out + i) = o;
}

// =====================================================================
// Kernel 1: fused Q/K/V projection.  msa_bf (32768 x 256) bf16 x W (256x256)
// Weight slices staged in LDS PRE-SWIZZLED into B-operand order.
// Q,K stored bf16 row-major [m][n];  V stored transposed [s][h][d][l] bf16.
// =====================================================================
__global__ __launch_bounds__(128) void qkv_proj_kernel(
    const __bf16* __restrict__ msa_bf,
    const float* __restrict__ Wq, const float* __restrict__ Wk,
    const float* __restrict__ Wv,
    __bf16* __restrict__ q_ws, __bf16* __restrict__ k_ws,
    __bf16* __restrict__ vt_ws)
{
  __shared__ __bf16 Wb[3][8][32][16];     // 24 KB, B-operand order
  const int n_base = blockIdx.y * 16;
  const int tid = threadIdx.x;
  for (int i = tid; i < 3 * DDIM * 16; i += 128) {
    int mat = i / (DDIM * 16);
    int rem = i % (DDIM * 16);
    int k = rem >> 4, nl = rem & 15;
    const float* W = (mat == 0) ? Wq : (mat == 1) ? Wk : Wv;
    int chunk = k >> 5, half = (k >> 4) & 1, e = k & 15;
    Wb[mat][chunk][half * 16 + nl][e] = f2bf(W[k * DDIM + n_base + nl]);
  }
  __syncthreads();

  const int wave = tid >> 5, lane = tid & 31;
  const int m_base = (blockIdx.x * 4 + wave) * 16;
  const int lrow  = lane & 15;
  const int koff8 = (lane >> 4) * 8;      // A-layout K split

  v8f aq = {}, ak = {}, av = {};
  const __bf16* arow = msa_bf + (size_t)(m_base + lrow) * DDIM;

#pragma unroll
  for (int c = 0; c < 8; ++c) {           // K chunks of 32
    v16bf a  = load8_8(arow + c * 32 + 2 * koff8, arow + c * 32 + 2 * koff8 + 8);
    v16bf bq = load16l(&Wb[0][c][lane][0]);
    v16bf bk = load16l(&Wb[1][c][lane][0]);
    v16bf bv = load16l(&Wb[2][c][lane][0]);
    aq = wmma_bf16(a, bq, aq);
    ak = wmma_bf16(a, bk, ak);
    av = wmma_bf16(a, bv, av);
  }

  const int n  = n_base + lrow;
  const int hh = n >> 5, dd = n & 31;
#pragma unroll
  for (int r = 0; r < 8; ++r) {
    int m = m_base + r + koff8;           // C-layout: M = r (+8 for upper half)
    int s = m >> 8, l = m & 255;
    q_ws[(size_t)m * DDIM + n] = f2bf(aq[r]);
    k_ws[(size_t)m * DDIM + n] = f2bf(ak[r]);
    vt_ws[(((size_t)s * NH + hh) * HD + dd) * LDIM + l] = f2bf(av[r]);
  }
}

// =====================================================================
// Kernel 2: pair bias  pb[h][l][m] = sum_p pair[l][m][p] * Wpb[p][h] (bf16 out)
// =====================================================================
__global__ __launch_bounds__(256) void pair_bias_kernel(
    const float* __restrict__ pair, const float* __restrict__ Wpb,
    __bf16* __restrict__ pb)
{
  const int l = blockIdx.y;
  const int m = blockIdx.x * 32 + (threadIdx.x >> 3);
  const int h = threadIdx.x & 7;
  const float* pr = pair + ((size_t)l * LDIM + m) * 128;
  float acc = 0.f;
  for (int p = 0; p < 128; ++p) acc += pr[p] * Wpb[p * NH + h];
  pb[((size_t)h * LDIM + l) * LDIM + m] = f2bf(acc);
}

// =====================================================================
// Kernel 3: attention.  Block = whole (s,h): 16 waves, one l-tile each.
// K staged once per block in LDS in A-operand order (16 KB), V^T staged in
// B-operand order (16 KB) -> 16x less L2 traffic than per-wave reloads.
// Scores computed transposed (C tile = (m,l)): softmax axis m lives in
// VGPR-index + lane-half -> one shfl_xor(16) per reduction.
// P staged per-wave in A-operand order (packed b128 stores), reloaded
// contiguously for the attn*V WMMAs.  Total LDS = 160 KB.
// =====================================================================
__global__ __launch_bounds__(512) void attn_kernel(
    const __bf16* __restrict__ q_ws, const __bf16* __restrict__ k_ws,
    const __bf16* __restrict__ vt_ws, const __bf16* __restrict__ pb,
    __bf16* __restrict__ o_ws)
{
  __shared__ __bf16 Kls[16][32][16];      // 16 KB, A-operand order
  __shared__ __bf16 Vls[8][2][32][16];    // 16 KB, B-operand order
  __shared__ __bf16 Pls[16][8][32][16];   // 128 KB, per-wave P (A-operand order)

  const int h = blockIdx.x;
  const int s = blockIdx.y;
  const int tid = threadIdx.x;
  const int wave = tid >> 5, lane = tid & 31;
  const int lrow   = lane & 15;
  const int koff8  = (lane >> 4) * 8;
  const int koff16 = (lane >> 4) * 16;
  const float scale = 0.17677669529663687f;   // 1/sqrt(32)

  // ---- cooperative staging: one 16-element slot per thread for K and V ----
  {
    int t = tid >> 5, ls = tid & 31;      // K slot (t, ls)
    int lr = ls & 15, k8 = (ls >> 4) * 8;
    const __bf16* kr = k_ws + ((size_t)(s * LDIM) + t * 16 + lr) * DDIM + h * HD;
    *(uint4*)&Kls[t][ls][0] = *(const uint4*)(kr + k8);
    *(uint4*)&Kls[t][ls][8] = *(const uint4*)(kr + 16 + k8);
  }
  {
    int j = tid >> 6, c = (tid >> 5) & 1, ls = tid & 31;   // V slot (j, c, ls)
    int d = c * 16 + (ls & 15), k8 = (ls >> 4) * 8;
    const __bf16* vr = vt_ws + ((size_t)s * NH + h) * HD * LDIM
                             + (size_t)d * LDIM + j * 32 + k8;
    *(uint4*)&Vls[j][c][ls][0] = *(const uint4*)vr;
    *(uint4*)&Vls[j][c][ls][8] = *(const uint4*)(vr + 16);
  }
  __syncthreads();

  const int l_base = wave * 16;

  // B operand (shared across all 16 score WMMAs): Q^T, N = l, K = d
  const __bf16* qrow =
      q_ws + ((size_t)(s * LDIM) + l_base + lrow) * DDIM + h * HD + koff8;
  const v16bf bq = load8_8(qrow, qrow + 16);

  v8f acc[16];
#pragma unroll
  for (int t = 0; t < 16; ++t) acc[t] = (v8f){};

#pragma unroll
  for (int t = 0; t < 16; ++t)            // m tiles: scores^T[m, l] = K . Q^T
    acc[t] = wmma_bf16(load16l(&Kls[t][lane][0]), bq, acc[t]);

  // scale + pair bias (bf16, one 16B load per tile), then exact softmax
  const __bf16* pbrow = pb + ((size_t)h * LDIM + l_base + lrow) * LDIM;
  float mx = -__builtin_inff();
#pragma unroll
  for (int t = 0; t < 16; ++t) {
    uint4 pw = *(const uint4*)(pbrow + t * 16 + koff8);
    float pbv[8] = { bflo(pw.x), bfhi(pw.x), bflo(pw.y), bfhi(pw.y),
                     bflo(pw.z), bfhi(pw.z), bflo(pw.w), bfhi(pw.w) };
#pragma unroll
    for (int r = 0; r < 8; ++r) {
      float v = acc[t][r] * scale + pbv[r];
      acc[t][r] = v;
      mx = fmaxf(mx, v);
    }
  }
  mx = fmaxf(mx, __shfl_xor(mx, 16, 32));

  float sum = 0.f;
#pragma unroll
  for (int t = 0; t < 16; ++t)
#pragma unroll
    for (int r = 0; r < 8; ++r) {
      float e = __expf(acc[t][r] - mx);
      acc[t][r] = e;
      sum += e;
    }
  sum += __shfl_xor(sum, 16, 32);
  const float inv = 1.f / sum;

  // stage P into LDS in A-operand order: one packed b128 store per t
#pragma unroll
  for (int t = 0; t < 16; ++t) {
    uint4 w;
    w.x = pack2(acc[t][0] * inv, acc[t][1] * inv);
    w.y = pack2(acc[t][2] * inv, acc[t][3] * inv);
    w.z = pack2(acc[t][4] * inv, acc[t][5] * inv);
    w.w = pack2(acc[t][6] * inv, acc[t][7] * inv);
    *(uint4*)&Pls[wave][t >> 1][lane][(t & 1) * 8] = w;
  }
  asm volatile("s_wait_dscnt 0" ::: "memory");   // same-wave LDS RAW guard

  // out[l, d] = sum_m P[l,m] V[m,d] ; A = P, B = V^T tiles (all from LDS)
  v8f o0 = {}, o1 = {};
#pragma unroll
  for (int j = 0; j < 8; ++j) {           // m chunks of 32
    v16bf ap = load16l(&Pls[wave][j][lane][0]);
    v16bf b0 = load16l(&Vls[j][0][lane][0]);   // d 0..15
    v16bf b1 = load16l(&Vls[j][1][lane][0]);   // d 16..31
    o0 = wmma_bf16(ap, b0, o0);
    o1 = wmma_bf16(ap, b1, o1);
  }

#pragma unroll
  for (int r = 0; r < 8; ++r) {
    int l = l_base + r + koff8;
    __bf16* orow = o_ws + ((size_t)(s * LDIM) + l) * DDIM + h * HD;
    orow[lrow]      = f2bf(o0[r]);
    orow[16 + lrow] = f2bf(o1[r]);
  }
}

// =====================================================================
// Kernel 4: final GEMM  out = attn_out(bf16) @ Wo  -> f32
// Wo staged in LDS pre-swizzled into B-operand order.
// =====================================================================
__global__ __launch_bounds__(128) void out_gemm_kernel(
    const __bf16* __restrict__ A, const float* __restrict__ Wo,
    float* __restrict__ out)
{
  __shared__ __bf16 Wb[8][32][16];        // 8 KB, B-operand order
  const int n_base = blockIdx.y * 16;
  const int tid = threadIdx.x;
  for (int i = tid; i < DDIM * 16; i += 128) {
    int k = i >> 4, nl = i & 15;
    int chunk = k >> 5, half = (k >> 4) & 1, e = k & 15;
    Wb[chunk][half * 16 + nl][e] = f2bf(Wo[k * DDIM + n_base + nl]);
  }
  __syncthreads();

  const int wave = tid >> 5, lane = tid & 31;
  const int m_base = (blockIdx.x * 4 + wave) * 16;
  const int lrow  = lane & 15;
  const int koff8 = (lane >> 4) * 8;

  v8f acc = {};
  const __bf16* arow = A + (size_t)(m_base + lrow) * DDIM;
#pragma unroll
  for (int c = 0; c < 8; ++c) {
    v16bf a = load8_8(arow + c * 32 + 2 * koff8, arow + c * 32 + 2 * koff8 + 8);
    v16bf b = load16l(&Wb[c][lane][0]);
    acc = wmma_bf16(a, b, acc);
  }
#pragma unroll
  for (int r = 0; r < 8; ++r)
    out[(size_t)(m_base + r + koff8) * DDIM + n_base + lrow] = acc[r];
}

// =====================================================================
extern "C" void kernel_launch(void* const* d_in, const int* in_sizes, int n_in,
                              void* d_out, int out_size, void* d_ws, size_t ws_size,
                              hipStream_t stream) {
  const float* msa  = (const float*)d_in[0];   // (1,128,256,256)
  const float* pair = (const float*)d_in[1];   // (1,256,256,128)
  const float* Wq   = (const float*)d_in[2];   // (256,256)
  const float* Wk   = (const float*)d_in[3];
  const float* Wv   = (const float*)d_in[4];
  const float* Wo   = (const float*)d_in[5];
  const float* Wpb  = (const float*)d_in[6];   // (128,8)
  float* out = (float*)d_out;                  // (1,128,256,256) f32

  const size_t M = (size_t)SDIM * LDIM;        // 32768 rows
  char* ws = (char*)d_ws;
  __bf16* msa_bf = (__bf16*)ws; ws += M * DDIM * sizeof(__bf16);            // 16 MB
  __bf16* q_ws   = (__bf16*)ws; ws += M * DDIM * sizeof(__bf16);            // 16 MB
  __bf16* k_ws   = (__bf16*)ws; ws += M * DDIM * sizeof(__bf16);            // 16 MB
  __bf16* vt_ws  = (__bf16*)ws; ws += (size_t)SDIM * NH * HD * LDIM * 2;    // 16 MB
  __bf16* o_ws   = (__bf16*)ws; ws += M * DDIM * sizeof(__bf16);            // 16 MB
  __bf16* pb     = (__bf16*)ws;                                             // 1 MB

  cvt_msa_kernel<<<dim3((unsigned)(M * DDIM / 8 / 256)), 256, 0, stream>>>(msa, msa_bf);
  qkv_proj_kernel<<<dim3(512, 16), 128, 0, stream>>>(msa_bf, Wq, Wk, Wv,
                                                     q_ws, k_ws, vt_ws);
  pair_bias_kernel<<<dim3(8, 256), 256, 0, stream>>>(pair, Wpb, pb);
  attn_kernel<<<dim3(NH, SDIM), 512, 0, stream>>>(q_ws, k_ws, vt_ws, pb, o_ws);
  out_gemm_kernel<<<dim3(512, 16), 128, 0, stream>>>(o_ws, Wo, out);
}
